// SAModule_24850680775352
// MI455X (gfx1250) — matrix operand order repaired and checked by reading, not compiled, not run on hardware
//
#include <hip/hip_runtime.h>
#include <hip/hip_bf16.h>

// ---------------------------------------------------------------------------
// PointNet++ SA module for MI455X (gfx1250, wave32).
//   B=8 clouds, N_PER=4096 pts, C_IN=64, M=1024 centers, K=64 nbrs, r=0.2
//   Kernel 1: sequential FPS per cloud (1 block/cloud).
//   Kernel 2: per-center ball query + gather + 3-layer MLP (WMMA f16) + max.
// ---------------------------------------------------------------------------

#define BB      8
#define N_PER   4096
#define C_IN    64
#define MM      1024
#define KK      64
#define R2      0.04f

typedef __attribute__((ext_vector_type(16))) _Float16 v16h;
typedef __attribute__((ext_vector_type(8)))  float    v8f;

// ------------------------------- FPS ---------------------------------------
// One block (256 threads) per cloud. Each thread owns 16 points in registers.
__global__ void fps_kernel(const float* __restrict__ pos,
                           float* __restrict__ pos_new,
                           float* __restrict__ batch_new,
                           int*   __restrict__ centers) {
    const int b   = blockIdx.x;
    const int tid = threadIdx.x;
    const float* P = pos + (size_t)b * N_PER * 3;
    const int PPT = N_PER / 256;          // 16 points per thread

    float px[PPT], py[PPT], pz[PPT], mind[PPT];
#pragma unroll
    for (int p = 0; p < PPT; ++p) {
        int j = tid * PPT + p;
        px[p] = P[j * 3 + 0];
        py[p] = P[j * 3 + 1];
        pz[p] = P[j * 3 + 2];
        mind[p] = 3.4e38f;
    }

    __shared__ int   idxs[MM];
    __shared__ float cur[3];
    __shared__ float rv[8];
    __shared__ int   ri[8];

    if (tid == 0) {
        idxs[0] = 0;
        cur[0] = P[0]; cur[1] = P[1]; cur[2] = P[2];
    }
    __syncthreads();

    for (int i = 1; i < MM; ++i) {
        const float cx = cur[0], cy = cur[1], cz = cur[2];
        float best = -1.0f;
        int   bi   = 0;
#pragma unroll
        for (int p = 0; p < PPT; ++p) {
            float dx = px[p] - cx, dy = py[p] - cy, dz = pz[p] - cz;
            float d2 = dx * dx + dy * dy + dz * dz;
            float m  = fminf(mind[p], d2);
            mind[p]  = m;
            if (m > best) { best = m; bi = tid * PPT + p; }
        }
        // wave32 argmax reduce (tie -> lowest index, matches jnp.argmax)
#pragma unroll
        for (int off = 16; off > 0; off >>= 1) {
            float ov = __shfl_xor(best, off);
            int   oi = __shfl_xor(bi,   off);
            if (ov > best || (ov == best && oi < bi)) { best = ov; bi = oi; }
        }
        const int wv = tid >> 5;
        if ((tid & 31) == 0) { rv[wv] = best; ri[wv] = bi; }
        __syncthreads();
        if (tid == 0) {
            float gb = rv[0]; int gi = ri[0];
#pragma unroll
            for (int k = 1; k < 8; ++k)
                if (rv[k] > gb || (rv[k] == gb && ri[k] < gi)) { gb = rv[k]; gi = ri[k]; }
            idxs[i] = gi;
            cur[0] = P[gi * 3 + 0]; cur[1] = P[gi * 3 + 1]; cur[2] = P[gi * 3 + 2];
        }
        __syncthreads();
    }

    // Emit centers (flat idx), pos_new, batch_new.
    for (int i = tid; i < MM; i += 256) {
        const int j    = idxs[i];
        const int flat = b * N_PER + j;
        const int o    = b * MM + i;
        centers[o] = flat;
        pos_new[o * 3 + 0] = P[j * 3 + 0];
        pos_new[o * 3 + 1] = P[j * 3 + 1];
        pos_new[o * 3 + 2] = P[j * 3 + 2];
        batch_new[o] = (float)b;
    }
}

// --------------------------- WMMA fragment loads ----------------------------
// A (16x32 f16): lane = M (lane&15); halves of K split by lane>=16 (+8),
// VGPR v<4 -> K = 2v, v>=4 -> K = 16+2(v-4); each VGPR packs K,K+1.
__device__ __forceinline__ v16h load_a_frag(const _Float16* H, int ld,
                                            int mbase, int kbase, int lane) {
    const int m  = mbase + (lane & 15);
    const int hi = (lane >> 4) & 1;
    v16h a;
#pragma unroll
    for (int v = 0; v < 8; ++v) {
        const int kb = (v < 4) ? (2 * v) : (16 + 2 * (v - 4));
        const int k  = kbase + kb + hi * 8;
        a[2 * v]     = H[m * ld + k];
        a[2 * v + 1] = H[m * ld + k + 1];
    }
    return a;
}

// B (32x16 f16): lane = column (lane&15); same K order as A: lanes>=16 +8,
// elements 0..7 -> K 0..7, 8..15 -> K 16..23 (Wt stored transposed [n][ld]).
__device__ __forceinline__ v16h load_b_frag(const _Float16* Wt, int ld,
                                            int nbase, int kbase, int lane) {
    const int n  = nbase + (lane & 15);
    const int k0 = kbase + ((lane >> 4) & 1) * 8;
    v16h bf;
#pragma unroll
    for (int e = 0; e < 16; ++e) bf[e] = Wt[n * ld + k0 + (e < 8 ? e : e + 8)];
    return bf;
}

// ------------------------------ SA kernel -----------------------------------
// 1 block (128 threads = 4 waves) per center.
__global__ void sa_kernel(const float* __restrict__ x,
                          const float* __restrict__ pos,
                          const float* __restrict__ W1, const float* __restrict__ b1,
                          const float* __restrict__ W2, const float* __restrict__ b2,
                          const float* __restrict__ W3, const float* __restrict__ b3,
                          const int*   __restrict__ centers,
                          float* __restrict__ xdest) {
    const int cIdx = blockIdx.x;                 // 0 .. B*M-1
    const int b    = cIdx / MM;
    const int bOff = b * N_PER;
    const int tid  = threadIdx.x;
    const int lane = tid & 31;
    const int w    = tid >> 5;                   // wave id 0..3 -> row tile

    // Phase-overlapped LDS carve (max concurrent footprint 40 KB).
    __shared__ __align__(16) char smem[40960];
    float*    d2s   = (float*)smem;                      // [0,16384)   phase A
    _Float16* Hs    = (_Float16*)smem;                   // [0,12288)   64x96
    _Float16* Wt1s  = (_Float16*)(smem + 12288);         // [12288,24576) 64x96
    _Float16* H2s   = (_Float16*)(smem + 24576);         // [24576,32768) 64x64
    _Float16* Wt2s  = (_Float16*)(smem + 32768);         // [32768,40960) 64x64
    _Float16* H3s   = (_Float16*)smem;                   // [0,8192)    64x64
    _Float16* Wt3s  = (_Float16*)(smem + 8192);          // [8192,24576) 128x64
    float*    colpart = (float*)(smem + 24576);          // [24576,26624) 4x128

    __shared__ float cpos[3];
    __shared__ int   nbr[KK];
    __shared__ int   cnt_s;

    const int flat = centers[cIdx];
    if (tid < 3) cpos[tid] = pos[(size_t)flat * 3 + tid];
    __syncthreads();
    const float cx = cpos[0], cy = cpos[1], cz = cpos[2];

    // ---- Phase A: squared distances to all 4096 points of this cloud ----
    for (int j = tid; j < N_PER; j += 128) {
        const float dx = pos[(size_t)(bOff + j) * 3 + 0] - cx;
        const float dy = pos[(size_t)(bOff + j) * 3 + 1] - cy;
        const float dz = pos[(size_t)(bOff + j) * 3 + 2] - cz;
        d2s[j] = dx * dx + dy * dy + dz * dz;
    }
    __syncthreads();

    // Count within radius.
    if (tid == 0) cnt_s = 0;
    __syncthreads();
    {
        int c = 0;
        for (int j = tid; j < N_PER; j += 128) c += (d2s[j] <= R2) ? 1 : 0;
        atomicAdd(&cnt_s, c);
    }
    __syncthreads();
    const int cAll = cnt_s;
    __syncthreads();

    // Binary-search the K-th distance boundary (count(t) <= K => deterministic).
    float t = R2;
    if (cAll > KK) {
        float lo = 0.0f, hi = R2;
        for (int it = 0; it < 20; ++it) {
            const float mid = 0.5f * (lo + hi);
            if (tid == 0) cnt_s = 0;
            __syncthreads();
            int c = 0;
            for (int j = tid; j < N_PER; j += 128) c += (d2s[j] <= mid) ? 1 : 0;
            atomicAdd(&cnt_s, c);
            __syncthreads();
            const int cm = cnt_s;
            __syncthreads();
            if (cm <= KK) lo = mid; else hi = mid;
        }
        t = lo;
    }

    // Compact neighbor indices (<= K of them; set is deterministic, order not —
    // irrelevant under max pooling).
    if (tid == 0) cnt_s = 0;
    __syncthreads();
    for (int j = tid; j < N_PER; j += 128) {
        if (d2s[j] <= t) {
            const int s = atomicAdd(&cnt_s, 1);
            if (s < KK) nbr[s] = j;
        }
    }
    __syncthreads();
    const int cnt = (cnt_s < KK) ? cnt_s : KK;
    __syncthreads();           // d2s dead; Hs/Wt1s may now overwrite it

    // ---- Phase B: stage H [64 x 96] (f16, zero-padded) + weights ----
    for (int e = tid; e < 64 * 96; e += 128) {
        const int s = e / 96, k = e % 96;
        _Float16 val = (_Float16)0.0f;
        if (s < cnt) {
            const int j = nbr[s];
            if (k < 64)      val = (_Float16)x[(size_t)(bOff + j) * C_IN + k];
            else if (k < 67) {
                const int c3 = k - 64;
                val = (_Float16)(pos[(size_t)(bOff + j) * 3 + c3] - cpos[c3]);
            }
        }
        Hs[e] = val;
    }
    for (int e = tid; e < 64 * 96; e += 128) {   // Wt1 [n][k], k<67 valid
        const int n = e / 96, k = e % 96;
        Wt1s[e] = (k < 67) ? (_Float16)W1[k * 64 + n] : (_Float16)0.0f;
    }
    for (int e = tid; e < 64 * 64; e += 128) {   // Wt2 [n][k]
        const int n = e / 64, k = e % 64;
        Wt2s[e] = (_Float16)W2[k * 64 + n];
    }
    __syncthreads();

    // ---- GEMM1: H[64,96] @ W1[96,64] + b1, ReLU -> H2 (f16) ----
    const int mbase = 16 * w;
    for (int ct = 0; ct < 4; ++ct) {
        const int nb = 16 * ct;
        const float bv = b1[nb + (lane & 15)];
        v8f acc;
#pragma unroll
        for (int r = 0; r < 8; ++r) acc[r] = bv;
#pragma unroll
        for (int kt = 0; kt < 3; ++kt) {
            const v16h a  = load_a_frag(Hs,   96, mbase, kt * 32, lane);
            const v16h bf = load_b_frag(Wt1s, 96, nb,    kt * 32, lane);
            acc = __builtin_amdgcn_wmma_f32_16x16x32_f16(false, a, false, bf,
                                                         (short)0, acc, false, false);
        }
        const int n  = nb + (lane & 15);
        const int hi = (lane >> 4) & 1;
#pragma unroll
        for (int r = 0; r < 8; ++r) {
            const int m = mbase + r + 8 * hi;
            float v = acc[r];
            H2s[m * 64 + n] = (_Float16)(v > 0.0f ? v : 0.0f);
        }
    }
    __syncthreads();

    // Stage Wt3 [128][64] into region that overlaps (now dead) Hs/Wt1s.
    for (int e = tid; e < 128 * 64; e += 128) {
        const int n = e / 64, k = e % 64;
        Wt3s[e] = (_Float16)W3[k * 128 + n];
    }

    // ---- GEMM2: H2[64,64] @ W2[64,64] + b2, ReLU -> H3 (f16) ----
    for (int ct = 0; ct < 4; ++ct) {
        const int nb = 16 * ct;
        const float bv = b2[nb + (lane & 15)];
        v8f acc;
#pragma unroll
        for (int r = 0; r < 8; ++r) acc[r] = bv;
#pragma unroll
        for (int kt = 0; kt < 2; ++kt) {
            const v16h a  = load_a_frag(H2s,  64, mbase, kt * 32, lane);
            const v16h bf = load_b_frag(Wt2s, 64, nb,    kt * 32, lane);
            acc = __builtin_amdgcn_wmma_f32_16x16x32_f16(false, a, false, bf,
                                                         (short)0, acc, false, false);
        }
        const int n  = nb + (lane & 15);
        const int hi = (lane >> 4) & 1;
#pragma unroll
        for (int r = 0; r < 8; ++r) {
            const int m = mbase + r + 8 * hi;
            float v = acc[r];
            H3s[m * 64 + n] = (_Float16)(v > 0.0f ? v : 0.0f);
        }
    }
    __syncthreads();

    // ---- GEMM3: H3[64,64] @ W3[64,128] + b3, mask invalid rows, max ----
    for (int ct = 0; ct < 8; ++ct) {
        const int nb = 16 * ct;
        const float bv = b3[nb + (lane & 15)];
        v8f acc;
#pragma unroll
        for (int r = 0; r < 8; ++r) acc[r] = bv;
#pragma unroll
        for (int kt = 0; kt < 2; ++kt) {
            const v16h a  = load_a_frag(H3s,  64, mbase, kt * 32, lane);
            const v16h bf = load_b_frag(Wt3s, 64, nb,    kt * 32, lane);
            acc = __builtin_amdgcn_wmma_f32_16x16x32_f16(false, a, false, bf,
                                                         (short)0, acc, false, false);
        }
        const int hi = (lane >> 4) & 1;
        float vm = -3.4e38f;
#pragma unroll
        for (int r = 0; r < 8; ++r) {
            const int srow = mbase + r + 8 * hi;          // neighbor slot
            const float v = (srow < cnt) ? acc[r] : -3.4e38f;
            vm = fmaxf(vm, v);
        }
        const float other = __shfl_xor(vm, 16);           // merge M halves
        vm = fmaxf(vm, other);
        if (lane < 16) colpart[w * 128 + nb + lane] = vm; // per-row-tile max
    }
    __syncthreads();

    for (int n = tid; n < 128; n += 128) {
        const float m01 = fmaxf(colpart[n],       colpart[128 + n]);
        const float m23 = fmaxf(colpart[256 + n], colpart[384 + n]);
        xdest[(size_t)cIdx * 128 + n] = fmaxf(m01, m23);
    }
}

// ------------------------------- launch -------------------------------------
extern "C" void kernel_launch(void* const* d_in, const int* in_sizes, int n_in,
                              void* d_out, int out_size, void* d_ws, size_t ws_size,
                              hipStream_t stream) {
    const float* x    = (const float*)d_in[0];
    const float* pos  = (const float*)d_in[1];
    // d_in[2] = batch (unused: derived analytically)
    const float* W1   = (const float*)d_in[3];
    const float* b1   = (const float*)d_in[4];
    const float* W2   = (const float*)d_in[5];
    const float* b2   = (const float*)d_in[6];
    const float* W3   = (const float*)d_in[7];
    const float* b3   = (const float*)d_in[8];

    float* out       = (float*)d_out;
    float* xdest     = out;                          // [B*M, 128]
    float* pos_new   = out + (size_t)BB * MM * 128;  // [B*M, 3]
    float* batch_new = pos_new + (size_t)BB * MM * 3;// [B*M]

    int* centers = (int*)d_ws;                       // B*M flat indices (32 KB)

    fps_kernel<<<BB, 256, 0, stream>>>(pos, pos_new, batch_new, centers);
    sa_kernel<<<BB * MM, 128, 0, stream>>>(x, pos, W1, b1, W2, b2, W3, b3,
                                           centers, xdest);
}
